// Lunsformter_37426345017802
// MI455X (gfx1250) — matrix operand order repaired and checked by reading, not compiled, not run on hardware
//
#include <hip/hip_runtime.h>
#include <hip/hip_bf16.h>
#include <math.h>

typedef _Float16 h16;
typedef __attribute__((ext_vector_type(16))) _Float16 v16h;
typedef __attribute__((ext_vector_type(8)))  float    v8f;
typedef __attribute__((ext_vector_type(4)))  unsigned int u32x4;
typedef __attribute__((ext_vector_type(2)))  unsigned int u32x2;

#define S_LEN  4096
#define DIM    1024
#define HID    4096
#define NLAYER 2
#define CHUNK  64
#define RDIM   256
#define WINSZ  8
#define VOCAB  32000

#define EPI_BIAS 0
#define EPI_GATE 1
#define EPI_TANH 2
#define EPI_ADD  3

// ---------------- f32 -> f16 conversion ----------------
__global__ void cvt_f32_f16(const float* __restrict__ in, h16* __restrict__ out, int n) {
  int i = blockIdx.x * blockDim.x + threadIdx.x;
  if (i < n) out[i] = (h16)in[i];
}

// ---------------- embed + positional + decay ----------------
__global__ void embed_kernel(const int* __restrict__ idx,
                             const float* __restrict__ emb,
                             const float* __restrict__ pos,
                             float* __restrict__ x, h16* __restrict__ xh) {
  int s = blockIdx.x;
  int d = blockIdx.y * blockDim.x + threadIdx.x;
  float v = emb[(size_t)idx[s] * DIM + d] + pos[(size_t)s * DIM + d];
  if (s > 0) {
    float x0 = emb[(size_t)idx[0] * DIM + d] + pos[d];
    float decay = powf(0.8f, (float)s);
    v += decay * x0;
  }
  x[(size_t)s * DIM + d] = v;
  xh[(size_t)s * DIM + d] = (h16)v;
}

// ---------------- WMMA GEMM, templated epilogue, 32-bit addressing ----------------
// Block: 256 threads = 8 waves in 4x2 grid; block tile 128(M) x 64(N); K staged 32-wide.
// Each wave: 32x32 subtile = 2x2 v_wmma_f32_16x16x32_f16.
template <int MODE, bool HAS_CF, bool HAS_CH>
__launch_bounds__(256)
__global__ void gemm_wmma_f16(const h16* __restrict__ A, const h16* __restrict__ B,
                              const float* __restrict__ bias,
                              float* __restrict__ Cf, h16* __restrict__ Ch,
                              const float* __restrict__ aux,
                              int M, int N, int K) {
  __shared__ unsigned int ldsA32[128 * 16];  // 128 rows x 32 f16 (K-pairs as dwords)
  __shared__ unsigned int ldsBp[64 * 16];    // 64 cols  x 16 K-pairs, pair-interleaved

  const int t    = threadIdx.x;
  const int wave = t >> 5;
  const int lane = t & 31;
  const int wm   = (wave >> 1) * 32;
  const int wn   = (wave & 1) * 32;
  const int m0   = blockIdx.y * 128;
  const int n0   = blockIdx.x * 64;
  const int laneHalf = lane >> 4;
  const int lm       = lane & 15;

  v8f acc[2][2];
  #pragma unroll
  for (int i = 0; i < 2; i++)
    #pragma unroll
    for (int j = 0; j < 2; j++)
      #pragma unroll
      for (int r = 0; r < 8; r++) acc[i][j][r] = 0.0f;

  // A staging: 512 uint4 units; unit = (row, kp4); 2 per thread
  const int arow0 = (t + 0)   >> 2, akp0 = (t + 0)   & 3;
  const int arow1 = (t + 256) >> 2, akp1 = (t + 256) & 3;
  // B staging: 256 uint2 units; unit = (k-pair p, n-quad q2); 1 per thread
  const int bp = t >> 4, bq2 = t & 15;

  for (int k0 = 0; k0 < K; k0 += 32) {
    __syncthreads();
    // --- stage A tile: 128x32 f16, global_load_b128 + ds_store_b128 ---
    {
      u32x4 a0 = *(const u32x4*)(A + (unsigned)((m0 + arow0) * K + k0 + akp0 * 8));
      u32x4 a1 = *(const u32x4*)(A + (unsigned)((m0 + arow1) * K + k0 + akp1 * 8));
      *(u32x4*)&ldsA32[t * 4]         = a0;
      *(u32x4*)&ldsA32[(t + 256) * 4] = a1;
    }
    // --- stage B tile 32x64, pair-interleaved: ldsBp[n*16+p] = {B[2p][n], B[2p+1][n]} ---
    {
      u32x2 u0 = *(const u32x2*)(B + (unsigned)((k0 + 2 * bp) * N + n0 + 4 * bq2));
      u32x2 u1 = *(const u32x2*)(B + (unsigned)((k0 + 2 * bp + 1) * N + n0 + 4 * bq2));
      ldsBp[(4 * bq2 + 0) * 16 + bp] = (u0.x & 0xffffu) | (u1.x << 16);
      ldsBp[(4 * bq2 + 1) * 16 + bp] = (u0.x >> 16) | (u1.x & 0xffff0000u);
      ldsBp[(4 * bq2 + 2) * 16 + bp] = (u0.y & 0xffffu) | (u1.y << 16);
      ldsBp[(4 * bq2 + 3) * 16 + bp] = (u0.y >> 16) | (u1.y & 0xffff0000u);
    }
    // prefetch next K slab while this one is consumed
    if (k0 + 32 < K) {
      __builtin_prefetch(A + (unsigned)((m0 + (t >> 1)) * K + k0 + 32), 0, 1);
      __builtin_prefetch(B + (unsigned)((k0 + 32 + (t >> 3)) * N + n0), 0, 1);
    }
    __syncthreads();

    // --- fragments: contiguous k = 16*laneHalf + s for A(16x32) and B(32x16) ---
    union { v16h v; unsigned int d[8]; } af[2], bf2[2];
    #pragma unroll
    for (int ti = 0; ti < 2; ti++) {
      int row = wm + ti * 16 + lm;      // M = lane%16
      #pragma unroll
      for (int v = 0; v < 8; v++) {
        int kp = v + (laneHalf << 3);
        af[ti].d[v] = ldsA32[row * 16 + kp];
      }
    }
    #pragma unroll
    for (int tj = 0; tj < 2; tj++) {
      int col = wn + tj * 16 + lm;      // N = lane%16
      #pragma unroll
      for (int v = 0; v < 8; v++)       // contiguous -> ds_load_b128
        bf2[tj].d[v] = ldsBp[col * 16 + v + (laneHalf << 3)];
    }

    #pragma unroll
    for (int ti = 0; ti < 2; ti++)
      #pragma unroll
      for (int tj = 0; tj < 2; tj++)
        acc[ti][tj] = __builtin_amdgcn_wmma_f32_16x16x32_f16(
            false, af[ti].v, false, bf2[tj].v, (short)0, acc[ti][tj], false, false);
  }

  // --- straight-line epilogue: vgpr r -> M = r + 8*laneHalf, N = lane%16 ---
  #pragma unroll
  for (int ti = 0; ti < 2; ti++) {
    #pragma unroll
    for (int tj = 0; tj < 2; tj++) {
      #pragma unroll
      for (int r = 0; r < 8; r++) {
        int m = m0 + wm + ti * 16 + laneHalf * 8 + r;
        int n = n0 + wn + tj * 16 + lm;
        unsigned o = (unsigned)(m * N + n);
        float v = acc[ti][tj][r] + bias[n];
        if constexpr (MODE == EPI_GATE) v = aux[o] * (1.0f / (1.0f + __expf(-v)));
        if constexpr (MODE == EPI_TANH) v = tanhf(v);
        if constexpr (MODE == EPI_ADD)  v = v + aux[o];
        if constexpr (HAS_CF) Cf[o] = v;
        if constexpr (HAS_CH) Ch[o] = (h16)v;
      }
    }
  }
}

// ---------------- windowed causal attention (wave per query row) ----------------
__launch_bounds__(256)
__global__ void attn_kernel(const float* __restrict__ Q, const float* __restrict__ Km,
                            const float* __restrict__ Vm, h16* __restrict__ aoh) {
  int wave = threadIdx.x >> 5;
  int lane = threadIdx.x & 31;
  int s = blockIdx.x * 8 + wave;

  float q[8];
  #pragma unroll
  for (int i = 0; i < 8; i++) q[i] = Q[(unsigned)(s * RDIM + lane * 8 + i)];

  float sc[WINSZ + 1];
  #pragma unroll
  for (int w = 0; w <= WINSZ; w++) {
    int j = s - w;
    float p = 0.0f;
    if (j >= 0) {                       // uniform per wave
      #pragma unroll
      for (int i = 0; i < 8; i++) p += q[i] * Km[(unsigned)(j * RDIM + lane * 8 + i)];
    }
    #pragma unroll
    for (int off = 16; off > 0; off >>= 1) p += __shfl_xor(p, off, 32);
    sc[w] = (j >= 0) ? p * (1.0f / 16.0f) : -1e30f;   // /sqrt(256)
  }
  float mx = sc[0];
  #pragma unroll
  for (int w = 1; w <= WINSZ; w++) mx = fmaxf(mx, sc[w]);
  float den = 0.0f;
  #pragma unroll
  for (int w = 0; w <= WINSZ; w++) { sc[w] = __expf(sc[w] - mx); den += sc[w]; }
  float inv = 1.0f / den;

  float o[8] = {0, 0, 0, 0, 0, 0, 0, 0};
  #pragma unroll
  for (int w = 0; w <= WINSZ; w++) {
    int j = s - w;
    if (j >= 0) {
      float a = sc[w] * inv;
      #pragma unroll
      for (int i = 0; i < 8; i++) o[i] += a * Vm[(unsigned)(j * RDIM + lane * 8 + i)];
    }
  }
  #pragma unroll
  for (int i = 0; i < 8; i++) aoh[(unsigned)(s * RDIM + lane * 8 + i)] = (h16)o[i];
}

// ---------------- chunk-link (elementwise, S % CHUNK == 0) ----------------
__global__ void chunk_kernel(const float* __restrict__ xin,
                             float* __restrict__ xout, h16* __restrict__ xh) {
  unsigned i = blockIdx.x * blockDim.x + threadIdx.x;
  int s = (int)(i / DIM);
  int c = s / CHUNK;
  const int n = S_LEN / CHUNK;
  float v = xin[i];
  float w = 1.0f;
  if (c > 0)     { v += 0.5f * xin[i - CHUNK * DIM]; w += 0.5f; }
  if (c < n - 1) { v += 0.5f * xin[i + CHUNK * DIM]; w += 0.5f; }
  v /= w;
  xout[i] = v;
  xh[i] = (h16)v;
}

// ---------------- host orchestration ----------------
static inline void* ws_alloc(char*& p, size_t bytes) {
  void* r = (void*)p;
  p += (bytes + 255) & ~(size_t)255;
  return r;
}

extern "C" void kernel_launch(void* const* d_in, const int* in_sizes, int n_in,
                              void* d_out, int out_size, void* d_ws, size_t ws_size,
                              hipStream_t stream) {
  const int*   idx  = (const int*)d_in[0];
  const float* emb  = (const float*)d_in[1];
  const float* pos  = (const float*)d_in[2];
  const float* Wg   = (const float*)d_in[3];
  const float* bg   = (const float*)d_in[4];
  const float* Ws   = (const float*)d_in[5];
  const float* bs   = (const float*)d_in[6];
  const float* Wo   = (const float*)d_in[7];
  const float* bo   = (const float*)d_in[8];
  const float* Wq   = (const float*)d_in[9];
  const float* Wk   = (const float*)d_in[10];
  const float* Wv   = (const float*)d_in[11];
  const float* Wao  = (const float*)d_in[12];
  const float* bq   = (const float*)d_in[13];
  const float* bk   = (const float*)d_in[14];
  const float* bv   = (const float*)d_in[15];
  const float* bao  = (const float*)d_in[16];
  const float* outW = (const float*)d_in[17];
  const float* outb = (const float*)d_in[18];
  float* logits = (float*)d_out;

  char* wsp = (char*)d_ws;
  h16* hWg   = (h16*)ws_alloc(wsp, (size_t)NLAYER * DIM * DIM * 2);
  h16* hWs   = (h16*)ws_alloc(wsp, (size_t)NLAYER * DIM * HID * 2);
  h16* hWo   = (h16*)ws_alloc(wsp, (size_t)NLAYER * HID * DIM * 2);
  h16* hWq   = (h16*)ws_alloc(wsp, (size_t)NLAYER * DIM * RDIM * 2);
  h16* hWk   = (h16*)ws_alloc(wsp, (size_t)NLAYER * DIM * RDIM * 2);
  h16* hWv   = (h16*)ws_alloc(wsp, (size_t)NLAYER * DIM * RDIM * 2);
  h16* hWao  = (h16*)ws_alloc(wsp, (size_t)NLAYER * RDIM * DIM * 2);
  h16* hOutW = (h16*)ws_alloc(wsp, (size_t)DIM * VOCAB * 2);
  float* x   = (float*)ws_alloc(wsp, (size_t)S_LEN * DIM * 4);
  h16*   xh  = (h16*)  ws_alloc(wsp, (size_t)S_LEN * DIM * 2);
  float* x1  = (float*)ws_alloc(wsp, (size_t)S_LEN * DIM * 4);
  h16*   x1h = (h16*)  ws_alloc(wsp, (size_t)S_LEN * DIM * 2);
  h16*   th  = (h16*)  ws_alloc(wsp, (size_t)S_LEN * HID * 2);
  float* x2  = (float*)ws_alloc(wsp, (size_t)S_LEN * DIM * 4);
  h16*   x2h = (h16*)  ws_alloc(wsp, (size_t)S_LEN * DIM * 2);
  float* Qf  = (float*)ws_alloc(wsp, (size_t)S_LEN * RDIM * 4);
  float* Kf  = (float*)ws_alloc(wsp, (size_t)S_LEN * RDIM * 4);
  float* Vf  = (float*)ws_alloc(wsp, (size_t)S_LEN * RDIM * 4);
  h16*   aoh = (h16*)  ws_alloc(wsp, (size_t)S_LEN * RDIM * 2);
  float* x3  = (float*)ws_alloc(wsp, (size_t)S_LEN * DIM * 4);
  (void)ws_size; (void)n_in; (void)in_sizes; (void)out_size;

  auto cvt = [&](const float* src, h16* dst, size_t n) {
    cvt_f32_f16<<<(unsigned)((n + 255) / 256), 256, 0, stream>>>(src, dst, (int)n);
  };
  cvt(Wg,   hWg,   (size_t)NLAYER * DIM * DIM);
  cvt(Ws,   hWs,   (size_t)NLAYER * DIM * HID);
  cvt(Wo,   hWo,   (size_t)NLAYER * HID * DIM);
  cvt(Wq,   hWq,   (size_t)NLAYER * DIM * RDIM);
  cvt(Wk,   hWk,   (size_t)NLAYER * DIM * RDIM);
  cvt(Wv,   hWv,   (size_t)NLAYER * DIM * RDIM);
  cvt(Wao,  hWao,  (size_t)NLAYER * RDIM * DIM);
  cvt(outW, hOutW, (size_t)DIM * VOCAB);

  embed_kernel<<<dim3(S_LEN, DIM / 256), 256, 0, stream>>>(idx, emb, pos, x, xh);

  for (int l = 0; l < NLAYER; l++) {
    // gate: x1 = x * sigmoid(x@Wg + bg)
    gemm_wmma_f16<EPI_GATE, true, true><<<dim3(DIM / 64, S_LEN / 128), 256, 0, stream>>>(
        xh, hWg + (size_t)l * DIM * DIM, bg + (size_t)l * DIM,
        x1, x1h, x, S_LEN, DIM, DIM);
    // sculpt part 1: th = tanh(x1@Ws + bs)
    gemm_wmma_f16<EPI_TANH, false, true><<<dim3(HID / 64, S_LEN / 128), 256, 0, stream>>>(
        x1h, hWs + (size_t)l * DIM * HID, bs + (size_t)l * HID,
        nullptr, th, nullptr, S_LEN, HID, DIM);
    // sculpt part 2 + residual: x2 = th@Wo + bo + x
    gemm_wmma_f16<EPI_ADD, true, true><<<dim3(DIM / 64, S_LEN / 128), 256, 0, stream>>>(
        th, hWo + (size_t)l * HID * DIM, bo + (size_t)l * DIM,
        x2, x2h, x, S_LEN, DIM, HID);
    // Q/K/V projections
    gemm_wmma_f16<EPI_BIAS, true, false><<<dim3(RDIM / 64, S_LEN / 128), 256, 0, stream>>>(
        x2h, hWq + (size_t)l * DIM * RDIM, bq + (size_t)l * RDIM,
        Qf, nullptr, nullptr, S_LEN, RDIM, DIM);
    gemm_wmma_f16<EPI_BIAS, true, false><<<dim3(RDIM / 64, S_LEN / 128), 256, 0, stream>>>(
        x2h, hWk + (size_t)l * DIM * RDIM, bk + (size_t)l * RDIM,
        Kf, nullptr, nullptr, S_LEN, RDIM, DIM);
    gemm_wmma_f16<EPI_BIAS, true, false><<<dim3(RDIM / 64, S_LEN / 128), 256, 0, stream>>>(
        x2h, hWv + (size_t)l * DIM * RDIM, bv + (size_t)l * RDIM,
        Vf, nullptr, nullptr, S_LEN, RDIM, DIM);
    // windowed attention
    attn_kernel<<<S_LEN / 8, 256, 0, stream>>>(Qf, Kf, Vf, aoh);
    // attn out + residual: x3 = aoh@Wao + bao + x2
    gemm_wmma_f16<EPI_ADD, true, false><<<dim3(DIM / 64, S_LEN / 128), 256, 0, stream>>>(
        aoh, hWao + (size_t)l * RDIM * DIM, bao + (size_t)l * DIM,
        x3, nullptr, x2, S_LEN, DIM, RDIM);
    // chunk link -> new x (f32 + f16)
    chunk_kernel<<<(S_LEN * DIM) / 256, 256, 0, stream>>>(x3, x, xh);
  }

  // final vocab projection straight into d_out
  gemm_wmma_f16<EPI_BIAS, true, false><<<dim3(VOCAB / 64, S_LEN / 128), 256, 0, stream>>>(
      xh, hOutW, outb, logits, nullptr, nullptr, S_LEN, VOCAB, DIM);
}
